// MultiHeadAttention_63728724738176
// MI455X (gfx1250) — compile-verified, matrix operand order checked by reading, not run on hardware
//
#include <hip/hip_runtime.h>

typedef __attribute__((ext_vector_type(16))) __bf16 v16bf;
typedef __attribute__((ext_vector_type(8)))  __bf16 v8bf;
typedef __attribute__((ext_vector_type(8)))  float  v8f;
typedef __attribute__((ext_vector_type(4)))  float  v4f;
typedef int v4i_gcc __attribute__((vector_size(16)));

namespace {
constexpr int kB  = 4;
constexpr int kN  = 4096;
constexpr int kD  = 1024;
constexpr int kH  = 16;
constexpr int kDK = 64;
constexpr int kM  = kB * kN;           // 16384 rows in the big GEMMs

// workspace layout (bytes)
constexpr size_t OFF_WQ  = 0;                       // 2 MB each (1024*1024 bf16)
constexpr size_t OFF_WK  = 2u  * 1024 * 1024;
constexpr size_t OFF_WV  = 4u  * 1024 * 1024;
constexpr size_t OFF_WO  = 6u  * 1024 * 1024;
constexpr size_t OFF_P   = 8u  * 1024 * 1024;       // 64 MB f32 projection / attn buffer
constexpr size_t OFF_QS  = 72u * 1024 * 1024;       // 32 MB bf16 softmaxed Q
constexpr size_t OFF_KT  = 104u * 1024 * 1024;      // 32 MB bf16 Kt [B,H,DK,N]
constexpr size_t OFF_VT  = 136u * 1024 * 1024;      // 32 MB bf16 Vt [B,H,DK,N]
constexpr size_t OFF_CTX = 168u * 1024 * 1024;      // 512 KB bf16 ctxT [B,H,DK,DK]
}

// ---- CDNA5 async copy helpers (guarded; fall back to sync LDS stores) ------
__device__ __forceinline__ void async_copy16(void* lds, const void* g) {
#if defined(__gfx1250__) && __has_builtin(__builtin_amdgcn_global_load_async_to_lds_b128)
  __builtin_amdgcn_global_load_async_to_lds_b128(
      (__attribute__((address_space(1))) v4i_gcc*)g,
      (__attribute__((address_space(3))) v4i_gcc*)lds, 0, 0);
#else
  *(int4*)lds = *(const int4*)g;
#endif
}

__device__ __forceinline__ void wait_async0() {
#if defined(__gfx1250__)
#if __has_builtin(__builtin_amdgcn_s_wait_asynccnt)
  __builtin_amdgcn_s_wait_asynccnt(0);
#else
  asm volatile("s_wait_asynccnt 0" ::: "memory");
#endif
#endif
}

// merge two 16-B LDS loads into a WMMA fragment (keeps 40-elem padded stride)
__device__ __forceinline__ v16bf ld_frag_lds(const __bf16* p) {
  v8bf lo = *(const v8bf*)p;
  v8bf hi = *(const v8bf*)(p + 8);
  return __builtin_shufflevector(lo, hi, 0, 1, 2, 3, 4, 5, 6, 7,
                                 8, 9, 10, 11, 12, 13, 14, 15);
}

__global__ void f32_to_bf16_kernel(const float* __restrict__ in,
                                   __bf16* __restrict__ out, int n) {
  int i = blockIdx.x * blockDim.x + threadIdx.x;
  if (i < n) out[i] = (__bf16)in[i];
}

// C[M,Nc] = A[M,K](f32, converted in-register) @ W[Nc,K]^T(bf16) + bias.
// Block = 4 waves sharing a 64-wide N strip; B tile double-buffered in LDS via
// async copies; A fragments register-double-buffered.
__global__ __launch_bounds__(128)
void gemm_f32a_wmma(const float* __restrict__ A, const __bf16* __restrict__ W,
                    const float* __restrict__ bias, float* __restrict__ C,
                    int M, int Nc, int K) {
  constexpr int BS = 40;                       // padded bf16 row stride
  __shared__ __bf16 Bs[2][64 * BS];

  const int tid = threadIdx.x;
  const int lane = tid & 31, wave = tid >> 5;
  const int lanelo = lane & 15, hi = lane >> 4;
  const int m0 = (blockIdx.x * 4 + wave) * 16;
  const int n0 = blockIdx.y * 64;

  // staging map: 64 rows x 64 B, 128 threads x 2 async b128 copies
  const int srow = tid >> 1, spart = tid & 1;
  const __bf16* wsrc = W + (size_t)(n0 + srow) * K + spart * 16;
  __bf16* sdst[2] = {&Bs[0][srow * BS + spart * 16], &Bs[1][srow * BS + spart * 16]};

  const float* aptr = A + (size_t)(m0 + lanelo) * K + hi * 16;

  auto stage = [&](int ks) {
    const __bf16* g = wsrc + ks * 32;
    __bf16* l = sdst[ks & 1];
    async_copy16(l, g);
    async_copy16(l + 8, g + 8);
  };

  stage(0);
  v4f a0 = *(const v4f*)(aptr + 0), a1 = *(const v4f*)(aptr + 4);
  v4f a2 = *(const v4f*)(aptr + 8), a3 = *(const v4f*)(aptr + 12);
  v4f na0 = a0, na1 = a1, na2 = a2, na3 = a3;
  wait_async0();
  __syncthreads();

  v8f acc0 = {}, acc1 = {}, acc2 = {}, acc3 = {};
  const int nk = K / 32;
  for (int ks = 0; ks < nk; ++ks) {
    if (ks + 1 < nk) {
      stage(ks + 1);                                  // async -> other buffer
      const float* ap = aptr + (ks + 1) * 32;
      __builtin_prefetch(ap + 32, 0, 1);
      na0 = *(const v4f*)(ap + 0);
      na1 = *(const v4f*)(ap + 4);
      na2 = *(const v4f*)(ap + 8);
      na3 = *(const v4f*)(ap + 12);
    }
    v16bf a;
#pragma unroll
    for (int i = 0; i < 4; ++i) {
      a[i]      = (__bf16)a0[i];
      a[4 + i]  = (__bf16)a1[i];
      a[8 + i]  = (__bf16)a2[i];
      a[12 + i] = (__bf16)a3[i];
    }
    const __bf16* bs = &Bs[ks & 1][lanelo * BS + hi * 16];
    v16bf b0 = ld_frag_lds(bs);
    v16bf b1 = ld_frag_lds(bs + 16 * BS);
    v16bf b2 = ld_frag_lds(bs + 32 * BS);
    v16bf b3 = ld_frag_lds(bs + 48 * BS);
    acc0 = __builtin_amdgcn_wmma_f32_16x16x32_bf16(false, a, false, b0, (short)0, acc0, false, false);
    acc1 = __builtin_amdgcn_wmma_f32_16x16x32_bf16(false, a, false, b1, (short)0, acc1, false, false);
    acc2 = __builtin_amdgcn_wmma_f32_16x16x32_bf16(false, a, false, b2, (short)0, acc2, false, false);
    acc3 = __builtin_amdgcn_wmma_f32_16x16x32_bf16(false, a, false, b3, (short)0, acc3, false, false);
    a0 = na0; a1 = na1; a2 = na2; a3 = na3;
    wait_async0();       // next stage landed
    __syncthreads();     // all waves done with this stage before it is reused
  }

  const float bb0 = bias[n0 + lanelo];
  const float bb1 = bias[n0 + 16 + lanelo];
  const float bb2 = bias[n0 + 32 + lanelo];
  const float bb3 = bias[n0 + 48 + lanelo];
#pragma unroll
  for (int r = 0; r < 8; ++r) {
    int m = m0 + hi * 8 + r;
    float* crow = C + (size_t)m * Nc + n0 + lanelo;
    crow[0]  = acc0[r] + bb0;
    crow[16] = acc1[r] + bb1;
    crow[32] = acc2[r] + bb2;
    crow[48] = acc3[r] + bb3;
  }
}

// softmax over last-dim 64-chunks, scale by 1/sqrt(64); f32 -> bf16
__global__ void q_softmax_kernel(const float* __restrict__ Qp,
                                 __bf16* __restrict__ Qs, int rows) {
  int r = blockIdx.x * blockDim.x + threadIdx.x;
  if (r >= rows) return;
  const float* x = Qp + (size_t)r * kDK;
  float m = -1e30f;
  for (int i = 0; i < kDK; ++i) m = fmaxf(m, x[i]);
  float s = 0.f;
  for (int i = 0; i < kDK; ++i) s += __expf(x[i] - m);
  float inv = 0.125f / s;  // 1/sqrt(dk) folded in
  __bf16* y = Qs + (size_t)r * kDK;
  for (int i = 0; i < kDK; ++i) y[i] = (__bf16)(__expf(x[i] - m) * inv);
}

// softmax over sequence dim per (b,h,d) column; writes transposed Kt[B,H,DK,N]
__global__ __launch_bounds__(256)
void k_softmax_T_kernel(const float* __restrict__ Kp, __bf16* __restrict__ Kt) {
  __shared__ float red[256];
  __shared__ float colmax[64];
  __shared__ float colsum[64];
  const int bh = blockIdx.x;             // b*H + h
  const int b = bh / kH, h = bh % kH;
  const int tid = threadIdx.x;
  const int d = tid & 63, j = tid >> 6;  // 64 columns x 4 partials
  const float* base = Kp + (size_t)b * kN * kD + h * kDK + d;

  float m = -1e30f;
  for (int n = j; n < kN; n += 4) m = fmaxf(m, base[(size_t)n * kD]);
  red[tid] = m;
  __syncthreads();
  if (j == 0)
    colmax[d] = fmaxf(fmaxf(red[d], red[64 + d]), fmaxf(red[128 + d], red[192 + d]));
  __syncthreads();
  const float gmax = colmax[d];

  float s = 0.f;
  for (int n = j; n < kN; n += 4) s += __expf(base[(size_t)n * kD] - gmax);
  __syncthreads();
  red[tid] = s;
  __syncthreads();
  if (j == 0) colsum[d] = red[d] + red[64 + d] + red[128 + d] + red[192 + d];
  __syncthreads();
  const float inv = 1.f / colsum[d];

  __bf16* out = Kt + ((size_t)bh * kDK + d) * kN;
  for (int n = j; n < kN; n += 4)
    out[n] = (__bf16)(__expf(base[(size_t)n * kD] - gmax) * inv);
}

// Vt[b,h,e,n] = bf16(Vp[b,n,h*64+e]) via 32x32 LDS tiles
__global__ __launch_bounds__(256)
void v_transpose_kernel(const float* __restrict__ Vp, __bf16* __restrict__ Vt) {
  __shared__ float tile[32][33];
  const int bh = blockIdx.x;
  const int b = bh / kH, h = bh % kH;
  const int n0 = blockIdx.y * 32, e0 = blockIdx.z * 32;
  const int tx = threadIdx.x, ty = threadIdx.y;
  for (int i = ty; i < 32; i += 8)
    tile[i][tx] = Vp[((size_t)b * kN + n0 + i) * kD + h * kDK + e0 + tx];
  __syncthreads();
  for (int i = ty; i < 32; i += 8)
    Vt[((size_t)bh * kDK + e0 + i) * kN + n0 + tx] = (__bf16)tile[tx][i];
}

// ctxT[b,h,e,d] = sum_n Kt[b,h,d,n] * Vt[b,h,e,n]; one wave per 16x16 tile,
// register double-buffered over the 4096-deep K loop.
__global__ __launch_bounds__(512)
void ctx_wmma_kernel(const __bf16* __restrict__ Kt, const __bf16* __restrict__ Vt,
                     __bf16* __restrict__ ctxT) {
  const int bh = blockIdx.x;
  const int lane = threadIdx.x & 31, wave = threadIdx.x >> 5;
  const int lanelo = lane & 15, hi = lane >> 4;
  const int d0 = (wave >> 2) * 16, e0 = (wave & 3) * 16;
  const __bf16* aptr = Kt + ((size_t)bh * kDK + d0 + lanelo) * kN + hi * 16;
  const __bf16* bptr = Vt + ((size_t)bh * kDK + e0 + lanelo) * kN + hi * 16;

  v16bf a = *(const v16bf*)aptr;
  v16bf b = *(const v16bf*)bptr;
  v8f acc = {};
  for (int k0 = 0; k0 < kN; k0 += 32) {
    v16bf an = a, bn = b;
    if (k0 + 32 < kN) {
      an = *(const v16bf*)(aptr + k0 + 32);
      bn = *(const v16bf*)(bptr + k0 + 32);
    }
    acc = __builtin_amdgcn_wmma_f32_16x16x32_bf16(false, a, false, b, (short)0, acc, false, false);
    a = an; b = bn;
  }
#pragma unroll
  for (int r = 0; r < 8; ++r) {
    int d = d0 + hi * 8 + r;
    int e = e0 + lanelo;
    ctxT[((size_t)bh * kDK + e) * kDK + d] = (__bf16)acc[r];
  }
}

// attn[b,n,h*64+e] = sum_d Qs[b,n,h*64+d] * ctx[d,e]; writes f32 for final GEMM
__global__ __launch_bounds__(256)
void attn_wmma_kernel(const __bf16* __restrict__ Qs, const __bf16* __restrict__ ctxT,
                      float* __restrict__ attn) {
  const int bh = blockIdx.x;
  const int b = bh / kH, h = bh % kH;
  const int lane = threadIdx.x & 31, wave = threadIdx.x >> 5;
  const int lanelo = lane & 15, hi = lane >> 4;
  const int n0 = blockIdx.y * 128 + wave * 16;

  const __bf16* aptr = Qs + ((size_t)b * kN + n0 + lanelo) * kD + h * kDK + hi * 16;
  const __bf16* cb   = ctxT + (size_t)bh * kDK * kDK + (size_t)lanelo * kDK + hi * 16;

  v8f acc0 = {}, acc1 = {}, acc2 = {}, acc3 = {};
#pragma unroll
  for (int k0 = 0; k0 < kDK; k0 += 32) {
    v16bf a  = *(const v16bf*)(aptr + k0);
    v16bf b0 = *(const v16bf*)(cb + 0 * 16 * kDK + k0);
    v16bf b1 = *(const v16bf*)(cb + 1 * 16 * kDK + k0);
    v16bf b2 = *(const v16bf*)(cb + 2 * 16 * kDK + k0);
    v16bf b3 = *(const v16bf*)(cb + 3 * 16 * kDK + k0);
    acc0 = __builtin_amdgcn_wmma_f32_16x16x32_bf16(false, a, false, b0, (short)0, acc0, false, false);
    acc1 = __builtin_amdgcn_wmma_f32_16x16x32_bf16(false, a, false, b1, (short)0, acc1, false, false);
    acc2 = __builtin_amdgcn_wmma_f32_16x16x32_bf16(false, a, false, b2, (short)0, acc2, false, false);
    acc3 = __builtin_amdgcn_wmma_f32_16x16x32_bf16(false, a, false, b3, (short)0, acc3, false, false);
  }
#pragma unroll
  for (int r = 0; r < 8; ++r) {
    int n = n0 + hi * 8 + r;
    float* orow = attn + ((size_t)b * kN + n) * kD + h * kDK;
    orow[0  + lanelo] = acc0[r];
    orow[16 + lanelo] = acc1[r];
    orow[32 + lanelo] = acc2[r];
    orow[48 + lanelo] = acc3[r];
  }
}

extern "C" void kernel_launch(void* const* d_in, const int* in_sizes, int n_in,
                              void* d_out, int out_size, void* d_ws, size_t ws_size,
                              hipStream_t stream) {
  (void)in_sizes; (void)n_in; (void)out_size; (void)ws_size;
  const float* query = (const float*)d_in[0];
  const float* key   = (const float*)d_in[1];
  const float* value = (const float*)d_in[2];
  const float* Wq = (const float*)d_in[3];
  const float* bq = (const float*)d_in[4];
  const float* Wk = (const float*)d_in[5];
  const float* bk = (const float*)d_in[6];
  const float* Wv = (const float*)d_in[7];
  const float* bv = (const float*)d_in[8];
  const float* Wo = (const float*)d_in[9];
  const float* bo = (const float*)d_in[10];

  char* ws = (char*)d_ws;
  __bf16* Wq_bf = (__bf16*)(ws + OFF_WQ);
  __bf16* Wk_bf = (__bf16*)(ws + OFF_WK);
  __bf16* Wv_bf = (__bf16*)(ws + OFF_WV);
  __bf16* Wo_bf = (__bf16*)(ws + OFF_WO);
  float*  bufP  = (float*)(ws + OFF_P);
  __bf16* Qs    = (__bf16*)(ws + OFF_QS);
  __bf16* Kt    = (__bf16*)(ws + OFF_KT);
  __bf16* Vt    = (__bf16*)(ws + OFF_VT);
  __bf16* ctxT  = (__bf16*)(ws + OFF_CTX);

  const int wn = kD * kD;
  const dim3 cvtW((wn + 255) / 256), blk(256);
  const dim3 ggrid(kM / 64, kD / 64);  // (256, 16)

  // weights -> bf16 (activations are consumed as f32 directly by the GEMM)
  f32_to_bf16_kernel<<<cvtW, blk, 0, stream>>>(Wq, Wq_bf, wn);
  f32_to_bf16_kernel<<<cvtW, blk, 0, stream>>>(Wk, Wk_bf, wn);
  f32_to_bf16_kernel<<<cvtW, blk, 0, stream>>>(Wv, Wv_bf, wn);
  f32_to_bf16_kernel<<<cvtW, blk, 0, stream>>>(Wo, Wo_bf, wn);

  // Q path
  gemm_f32a_wmma<<<ggrid, 128, 0, stream>>>(query, Wq_bf, bq, bufP, kM, kD, kD);
  q_softmax_kernel<<<(kB * kN * kH + 255) / 256, blk, 0, stream>>>(bufP, Qs, kB * kN * kH);

  // K path (softmax over N, writes transposed)
  gemm_f32a_wmma<<<ggrid, 128, 0, stream>>>(key, Wk_bf, bk, bufP, kM, kD, kD);
  k_softmax_T_kernel<<<kB * kH, 256, 0, stream>>>(bufP, Kt);

  // V path (transpose to feature-major)
  gemm_f32a_wmma<<<ggrid, 128, 0, stream>>>(value, Wv_bf, bv, bufP, kM, kD, kD);
  v_transpose_kernel<<<dim3(kB * kH, kN / 32, kDK / 32), dim3(32, 8), 0, stream>>>(bufP, Vt);

  // ctx = K^T V (per head 64x4096x64)
  ctx_wmma_kernel<<<kB * kH, 512, 0, stream>>>(Kt, Vt, ctxT);

  // out = Q ctx (per head 4096x64x64) -> f32 into bufP
  attn_wmma_kernel<<<dim3(kB * kH, kN / 128), 256, 0, stream>>>(Qs, ctxT, bufP);

  // final projection -> f32 d_out
  gemm_f32a_wmma<<<ggrid, 128, 0, stream>>>(bufP, Wo_bf, bo, (float*)d_out, kM, kD, kD);
}
